// VQEmbedding_13950053777635
// MI455X (gfx1250) — matrix operand order repaired and checked by reading, not compiled, not run on hardware
//
#include <hip/hip_runtime.h>
#include <hip/hip_bf16.h>

// ---------- types ----------
typedef __attribute__((ext_vector_type(16))) __bf16 v16bf;
typedef __attribute__((ext_vector_type(8)))  float  v8f;
typedef __attribute__((ext_vector_type(8)))  unsigned int v8u;

struct U32x8 { uint4 lo, hi; };   // 32 bytes, bit-castable to v16bf

#define D_DIM 256
#define WSTR  264                 // padded LDS row stride in halves (16B aligned, bank-spread)

// ---------- bf16 helpers (round-to-nearest-even) ----------
static __device__ inline unsigned short f2bf(float f) {
    unsigned int u = __float_as_uint(f);
    u += 0x7FFFu + ((u >> 16) & 1u);
    return (unsigned short)(u >> 16);
}
static __device__ inline float bf2f(unsigned short h) {
    return __uint_as_float(((unsigned int)h) << 16);
}
static __device__ inline void split2(float x, unsigned short& hb, unsigned short& lb) {
    hb = f2bf(x);
    lb = f2bf(x - bf2f(hb));
}

// ---------- CDNA5 async global->LDS DMA (no VGPR data path, ASYNCcnt) ----------
static __device__ inline void async_ld_b128(unsigned lds_off, unsigned goff,
                                            unsigned long long base) {
    asm volatile("global_load_async_to_lds_b128 %0, %1, %2"
                 :: "v"(lds_off), "v"(goff), "s"(base) : "memory");
}
static __device__ inline void wait_async0() {
    asm volatile("s_wait_asynccnt 0x0" ::: "memory");
}

// =====================================================================
// Kernel 1: w_sqr[k] = sum_d W[k,d]^2  (wave per code row), and when
// PRESPLIT: also emit bf16 hi/lo planes of W for the GEMM hot loop.
// =====================================================================
template <bool PRESPLIT>
__global__ __launch_bounds__(256) void vq_wsqr_kernel(const float* __restrict__ w,
                                                      float* __restrict__ wsq,
                                                      unsigned short* __restrict__ whi,
                                                      unsigned short* __restrict__ wlo,
                                                      int K) {
    const int lane = threadIdx.x & 31;
    const int wave = threadIdx.x >> 5;
    const int k = blockIdx.x * 8 + wave;
    if (k >= K) return;
    const float* wr = w + (size_t)k * D_DIM;

    float4 f0 = *(const float4*)(wr + lane * 8);
    float4 f1 = *(const float4*)(wr + lane * 8 + 4);
    float ff[8] = {f0.x, f0.y, f0.z, f0.w, f1.x, f1.y, f1.z, f1.w};

    float s = 0.f;
#pragma unroll
    for (int j = 0; j < 8; ++j) s += ff[j] * ff[j];

    if constexpr (PRESPLIT) {
        unsigned short hb[8], lb[8];
#pragma unroll
        for (int j = 0; j < 8; ++j) split2(ff[j], hb[j], lb[j]);
        uint4 uh, ul;
        uh.x = (unsigned)hb[0] | ((unsigned)hb[1] << 16);
        uh.y = (unsigned)hb[2] | ((unsigned)hb[3] << 16);
        uh.z = (unsigned)hb[4] | ((unsigned)hb[5] << 16);
        uh.w = (unsigned)hb[6] | ((unsigned)hb[7] << 16);
        ul.x = (unsigned)lb[0] | ((unsigned)lb[1] << 16);
        ul.y = (unsigned)lb[2] | ((unsigned)lb[3] << 16);
        ul.z = (unsigned)lb[4] | ((unsigned)lb[5] << 16);
        ul.w = (unsigned)lb[6] | ((unsigned)lb[7] << 16);
        *(uint4*)(whi + (size_t)k * D_DIM + lane * 8) = uh;
        *(uint4*)(wlo + (size_t)k * D_DIM + lane * 8) = ul;
    }

#pragma unroll
    for (int off = 16; off > 0; off >>= 1) s += __shfl_xor(s, off, 32);
    if (lane == 0) wsq[k] = s;
}

// =====================================================================
// Kernel 2: per-row argmin over K codes of (||w||^2 - 2 z.w)
//   block = 256 threads = 8 waves; wave = 16 rows; block = 128 rows.
//   bf16x3 split GEMM on v_wmma_f32_16x16x32_bf16 with 3 independent
//   accumulator chains; double-buffered LDS W tiles.
//   PRESPLIT: tiles staged by global_load_async_to_lds_b128 (ASYNCcnt),
//   zero staging VGPRs. Fallback: register-staged + in-loop bf16 split.
// =====================================================================
template <bool PRESPLIT>
__global__ __launch_bounds__(256) void vq_argmin_kernel(const float* __restrict__ z,
                                                        const float* __restrict__ w,
                                                        const unsigned short* __restrict__ whi,
                                                        const unsigned short* __restrict__ wlo,
                                                        const float* __restrict__ wsq,
                                                        int* __restrict__ idxOut, int K) {
    __shared__ __align__(16) unsigned short lds_whi[2][16 * WSTR];
    __shared__ __align__(16) unsigned short lds_wlo[2][16 * WSTR];

    const int tid     = threadIdx.x;
    const int lane    = tid & 31;
    const int wave    = tid >> 5;
    const int lane16  = lane & 15;
    const int halfsel = lane >> 4;                 // 0: lanes 0-15, 1: lanes 16-31
    const int rowBase = blockIdx.x * 128 + wave * 16;
    const int r       = rowBase + lane16;          // A-matrix row for this lane

    // LDS byte offsets of the double buffers (generic ptr low 32 bits = LDS addr)
    const unsigned ldsH0 = (unsigned)(uintptr_t)&lds_whi[0][0];
    const unsigned ldsH1 = (unsigned)(uintptr_t)&lds_whi[1][0];
    const unsigned ldsL0 = (unsigned)(uintptr_t)&lds_wlo[0][0];
    const unsigned ldsL1 = (unsigned)(uintptr_t)&lds_wlo[1][0];
    const unsigned long long whiB = (unsigned long long)(uintptr_t)whi;
    const unsigned long long wloB = (unsigned long long)(uintptr_t)wlo;

    // this thread's staging chunk coordinates (2 x b128 per plane per tile)
    const int g0 = tid,        row0 = g0 >> 5, c0i = g0 & 31;
    const int g1 = 256 + tid,  row1 = g1 >> 5, c1i = g1 & 31;
    const unsigned tileOff0 = (unsigned)((row0 * WSTR + c0i * 8) * 2);  // LDS bytes
    const unsigned tileOff1 = (unsigned)((row1 * WSTR + c1i * 8) * 2);

    // ---- A tile: 16x256 z rows -> registers, bf16 hi/lo split.
    // A k order used here (contiguous, matched by B): lanes<16 hold K 0-15 of each 32-slice,
    // lanes>=16 hold K 16-31.
    v16bf a_hi[8], a_lo[8];
    {
        const float* zr = z + (size_t)r * D_DIM;
#pragma unroll
        for (int s = 0; s < 8; ++s) {
            v8u phi, plo;
            const int c0 = s * 32 + halfsel * 16;
#pragma unroll
            for (int h = 0; h < 2; ++h) {
                const int cb = c0 + h * 8;
                float4 f0 = *(const float4*)(zr + cb);
                float4 f1 = *(const float4*)(zr + cb + 4);
                float ff[8] = {f0.x, f0.y, f0.z, f0.w, f1.x, f1.y, f1.z, f1.w};
#pragma unroll
                for (int j = 0; j < 4; ++j) {
                    unsigned short h0, l0, h1, l1;
                    split2(ff[2 * j],     h0, l0);
                    split2(ff[2 * j + 1], h1, l1);
                    phi[h * 4 + j] = (unsigned)h0 | ((unsigned)h1 << 16);
                    plo[h * 4 + j] = (unsigned)l0 | ((unsigned)l1 << 16);
                }
            }
            a_hi[s] = __builtin_bit_cast(v16bf, phi);
            a_lo[s] = __builtin_bit_cast(v16bf, plo);
        }
    }

    float minv[8];
    int   mini[8];
#pragma unroll
    for (int v = 0; v < 8; ++v) { minv[v] = 3.402823466e38f; mini[v] = 0; }

    const int numTiles = K >> 4;

    // ---- prologue: stage tile 0 into buffer 0
    if constexpr (PRESPLIT) {
        const unsigned gb0 = (unsigned)((row0 * D_DIM + c0i * 8) * 2);
        const unsigned gb1 = (unsigned)((row1 * D_DIM + c1i * 8) * 2);
        async_ld_b128(ldsH0 + tileOff0, gb0, whiB);
        async_ld_b128(ldsH0 + tileOff1, gb1, whiB);
        async_ld_b128(ldsL0 + tileOff0, gb0, wloB);
        async_ld_b128(ldsL0 + tileOff1, gb1, wloB);
    } else {
#pragma unroll
        for (int i = 0; i < 4; ++i) {
            const int g = i * 256 + tid;           // float4 group 0..1023
            const int row = g >> 6, d4 = g & 63;
            float4 f = *(const float4*)(w + (size_t)row * D_DIM + d4 * 4);
            unsigned short h0, l0, h1, l1, h2, l2, h3, l3;
            split2(f.x, h0, l0); split2(f.y, h1, l1);
            split2(f.z, h2, l2); split2(f.w, h3, l3);
            uint2 ph, pl;
            ph.x = (unsigned)h0 | ((unsigned)h1 << 16);
            ph.y = (unsigned)h2 | ((unsigned)h3 << 16);
            pl.x = (unsigned)l0 | ((unsigned)l1 << 16);
            pl.y = (unsigned)l2 | ((unsigned)l3 << 16);
            const int off = row * WSTR + d4 * 4;
            *(uint2*)&lds_whi[0][off] = ph;
            *(uint2*)&lds_wlo[0][off] = pl;
        }
    }

    for (int t = 0; t < numTiles; ++t) {
        const int sel = t & 1;
        const bool have_next = (t + 1 < numTiles);

        if constexpr (PRESPLIT) {
            wait_async0();                         // my tile-t DMAs landed in LDS
        }
        __syncthreads();                           // all waves staged t; old readers of buf[(t+1)&1] done

        // ---- stage next tile into alternate buffer (PRESPLIT: async DMA,
        // proceeds underneath the WMMA block; no VGPRs touched)
        float4 nf[4];                              // fallback staging regs only
        if (have_next) {
            if constexpr (PRESPLIT) {
                const unsigned nH = sel ? ldsH0 : ldsH1;
                const unsigned nL = sel ? ldsL0 : ldsL1;
                const unsigned tb = (unsigned)((t + 1) * 16 * D_DIM * 2);
                const unsigned gb0 = tb + (unsigned)((row0 * D_DIM + c0i * 8) * 2);
                const unsigned gb1 = tb + (unsigned)((row1 * D_DIM + c1i * 8) * 2);
                async_ld_b128(nH + tileOff0, gb0, whiB);
                async_ld_b128(nH + tileOff1, gb1, whiB);
                async_ld_b128(nL + tileOff0, gb0, wloB);
                async_ld_b128(nL + tileOff1, gb1, wloB);
            } else {
#pragma unroll
                for (int i = 0; i < 4; ++i) {
                    const int g = i * 256 + tid;
                    const int row = g >> 6, d4 = g & 63;
                    nf[i] = *(const float4*)(w + (size_t)((t + 1) * 16 + row) * D_DIM + d4 * 4);
                }
            }
            if (t + 2 < numTiles) {                // keep-near prefetch two tiles ahead
                if constexpr (PRESPLIT) {
                    __builtin_prefetch((const void*)(whi + (size_t)(t + 2) * 16 * D_DIM + (tid << 4)), 0, 3);
                    __builtin_prefetch((const void*)(wlo + (size_t)(t + 2) * 16 * D_DIM + (tid << 4)), 0, 3);
                } else {
                    __builtin_prefetch((const void*)(w + (size_t)(t + 2) * 16 * D_DIM + (tid << 4)), 0, 3);
                }
            }
        }

        const unsigned short* Lhi = &lds_whi[sel][0];
        const unsigned short* Llo = &lds_wlo[sel][0];
        const float ws = wsq[t * 16 + lane16];

        // ---- 24 WMMAs, 3 independent accumulator chains (8-deep each)
        v8f acc0 = {}, acc1 = {}, acc2 = {};
#pragma unroll
        for (int s = 0; s < 8; ++s) {
            const int k0 = s * 32 + halfsel * 16;  // B layout: lanes<16 K 0-15, lanes>=16 K 16-31
            const unsigned short* bph = Lhi + lane16 * WSTR + k0;
            const unsigned short* bpl = Llo + lane16 * WSTR + k0;
            U32x8 th, tl;
            th.lo = *(const uint4*)bph; th.hi = *(const uint4*)(bph + 8);
            tl.lo = *(const uint4*)bpl; tl.hi = *(const uint4*)(bpl + 8);
            v16bf bhi = __builtin_bit_cast(v16bf, th);
            v16bf blo = __builtin_bit_cast(v16bf, tl);
            acc0 = __builtin_amdgcn_wmma_f32_16x16x32_bf16(false, a_hi[s], false, bhi,
                                                           (short)0, acc0, false, false);
            acc1 = __builtin_amdgcn_wmma_f32_16x16x32_bf16(false, a_hi[s], false, blo,
                                                           (short)0, acc1, false, false);
            acc2 = __builtin_amdgcn_wmma_f32_16x16x32_bf16(false, a_lo[s], false, bhi,
                                                           (short)0, acc2, false, false);
        }

        // ---- running min/argmin (C layout: M = v + 8*halfsel, N = lane16)
        const int n = t * 16 + lane16;
#pragma unroll
        for (int v = 0; v < 8; ++v) {
            float dot = acc0[v] + acc1[v] + acc2[v];
            float score = ws - 2.0f * dot;
            if (score < minv[v]) { minv[v] = score; mini[v] = n; }
        }

        // ---- fallback path: convert + store next tile after compute
        if constexpr (!PRESPLIT) {
            if (have_next) {
                const int nsel = sel ^ 1;
#pragma unroll
                for (int i = 0; i < 4; ++i) {
                    const int g = i * 256 + tid;
                    const int row = g >> 6, d4 = g & 63;
                    unsigned short h0, l0, h1, l1, h2, l2, h3, l3;
                    split2(nf[i].x, h0, l0); split2(nf[i].y, h1, l1);
                    split2(nf[i].z, h2, l2); split2(nf[i].w, h3, l3);
                    uint2 ph, pl;
                    ph.x = (unsigned)h0 | ((unsigned)h1 << 16);
                    ph.y = (unsigned)h2 | ((unsigned)h3 << 16);
                    pl.x = (unsigned)l0 | ((unsigned)l1 << 16);
                    pl.y = (unsigned)l2 | ((unsigned)l3 << 16);
                    const int off = row * WSTR + d4 * 4;
                    *(uint2*)&lds_whi[nsel][off] = ph;
                    *(uint2*)&lds_wlo[nsel][off] = pl;
                }
            }
        }
    }

    // ---- reduce each row's 16 N-lanes (width-16 xor shuffle), first-index ties
#pragma unroll
    for (int v = 0; v < 8; ++v) {
        float mv = minv[v];
        int   mi = mini[v];
#pragma unroll
        for (int off = 8; off > 0; off >>= 1) {
            float ov = __shfl_xor(mv, off, 16);
            int   oi = __shfl_xor(mi, off, 16);
            if (ov < mv || (ov == mv && oi < mi)) { mv = ov; mi = oi; }
        }
        if (lane16 == 0) idxOut[rowBase + v + halfsel * 8] = mi;
    }
}

// =====================================================================
// Kernel 3: exact fp32 loss  out[b] = 1.25 * sum_d (z[b,d]-W[k*,d])^2
// =====================================================================
__global__ __launch_bounds__(256) void vq_loss_kernel(const float* __restrict__ z,
                                                      const float* __restrict__ w,
                                                      const int* __restrict__ idx,
                                                      float* __restrict__ out, int B) {
    const int lane = threadIdx.x & 31;
    const int wave = threadIdx.x >> 5;
    const int b = blockIdx.x * 8 + wave;
    if (b >= B) return;
    const int k = idx[b];
    const float* zr = z + (size_t)b * D_DIM;
    const float* wr = w + (size_t)k * D_DIM;
    float s = 0.f;
#pragma unroll
    for (int i = 0; i < 2; ++i) {
        float4 a = *(const float4*)(zr + lane * 8 + i * 4);
        float4 c = *(const float4*)(wr + lane * 8 + i * 4);
        float dx = a.x - c.x, dy = a.y - c.y, dz = a.z - c.z, dw = a.w - c.w;
        s += dx * dx + dy * dy + dz * dz + dw * dw;
    }
#pragma unroll
    for (int off = 16; off > 0; off >>= 1) s += __shfl_xor(s, off, 32);
    if (lane == 0) out[b] = 1.25f * s;             // (1 + BETA), BETA = 0.25
}

// =====================================================================
extern "C" void kernel_launch(void* const* d_in, const int* in_sizes, int n_in,
                              void* d_out, int out_size, void* d_ws, size_t ws_size,
                              hipStream_t stream) {
    const float* z = (const float*)d_in[0];        // [B, 256]
    const float* w = (const float*)d_in[1];        // [K, 256]
    const int B = in_sizes[0] / D_DIM;             // 16384
    const int K = in_sizes[1] / D_DIM;             // 4096

    // workspace layout
    const size_t off_wsq = 0;                                  // K floats
    const size_t off_idx = (size_t)K * sizeof(float);          // B ints
    const size_t off_whi = off_idx + (size_t)B * sizeof(int);  // K*256 halves
    const size_t off_wlo = off_whi + (size_t)K * D_DIM * 2;
    const size_t needed  = off_wlo + (size_t)K * D_DIM * 2;

    float*          wsq = (float*)((char*)d_ws + off_wsq);
    int*            idx = (int*)((char*)d_ws + off_idx);
    unsigned short* whi = (unsigned short*)((char*)d_ws + off_whi);
    unsigned short* wlo = (unsigned short*)((char*)d_ws + off_wlo);
    float*          out = (float*)d_out;

    const bool presplit = (ws_size >= needed);     // constant per run -> deterministic

    if (presplit) {
        vq_wsqr_kernel<true><<<(K + 7) / 8, 256, 0, stream>>>(w, wsq, whi, wlo, K);
        vq_argmin_kernel<true><<<B / 128, 256, 0, stream>>>(z, w, whi, wlo, wsq, idx, K);
    } else {
        vq_wsqr_kernel<false><<<(K + 7) / 8, 256, 0, stream>>>(w, wsq, nullptr, nullptr, K);
        vq_argmin_kernel<false><<<B / 128, 256, 0, stream>>>(z, w, nullptr, nullptr, wsq, idx, K);
    }
    vq_loss_kernel<<<(B + 7) / 8, 256, 0, stream>>>(z, w, idx, out, B);
}
